// RandomWalkAttention_84301618086509
// MI455X (gfx1250) — matrix operand order repaired and checked by reading, not compiled, not run on hardware
//
#include <hip/hip_runtime.h>
#include <hip/hip_bf16.h>

typedef __attribute__((ext_vector_type(16))) _Float16 v16h;
typedef __attribute__((ext_vector_type(8)))  float    v8f;

#define B_SZ   128
#define NW_SZ  64
#define LW_SZ  32
#define TD_SZ  108
#define PD_SZ  19
#define F_DIM  128
#define M_DIM  128
#define RSQRT128 0.088388347648318447f   // 1/sqrt(128)

// ---------------- WMMA fragment helpers (CDNA5 16x16x32 f16, wave32) -------
// A-matrix 16x32 (MxK), lane<16: M=lane, K in {k0..k0+7, k0+16..k0+23};
// lane>=16: M=lane-16, K in {k0+8..k0+15, k0+24..k0+31}  (ISA 7.12.2)
__device__ __forceinline__ v16h load_a_frag(const _Float16* src, int ld,
                                            int m0, int k0, int lane) {
    const int hi16 = (lane >> 4) & 1;
    const _Float16* row = src + (m0 + (lane & 15)) * ld;
    const int klo = k0 + hi16 * 8;
    const int khi = k0 + 16 + hi16 * 8;
    v16h a;
#pragma unroll
    for (int j = 0; j < 8; ++j) a[j] = row[klo + j];
#pragma unroll
    for (int j = 0; j < 8; ++j) a[8 + j] = row[khi + j];
    return a;
}

// B-matrix 32x16 (KxN); element (k,n) = src[n*ld + k]  (src rows indexed by N,
// contiguous in K).  lane<16: N=lane, K in {k0..k0+7, k0+16..k0+23};
// lane>=16: N=lane-16, K in {k0+8..k0+15, k0+24..k0+31}.
__device__ __forceinline__ v16h load_b_frag_nk(const _Float16* src, int ld,
                                               int k0, int n0, int lane) {
    const int kb = k0 + ((lane >> 4) & 1) * 8;
    const _Float16* row = src + (n0 + (lane & 15)) * ld;
    v16h b;
#pragma unroll
    for (int i = 0; i < 16; ++i) b[i] = row[kb + (i & 7) + (i >> 3) * 16];
    return b;
}

// C/D 16x16 f32: VGPR r holds (M = m0 + r + 8*(lane>=16), N = n0 + lane%16)
__device__ __forceinline__ void store_c_f16(_Float16* dst, int ld, int m0, int n0,
                                            int lane, v8f c) {
    const int n  = n0 + (lane & 15);
    const int mb = m0 + ((lane >> 4) & 1) * 8;
#pragma unroll
    for (int r = 0; r < 8; ++r) dst[(mb + r) * ld + n] = (_Float16)c[r];
}

__device__ __forceinline__ void store_c_f32(float* dst, int ld, int m0, int n0,
                                            int lane, v8f c, float scale) {
    const int n  = n0 + (lane & 15);
    const int mb = m0 + ((lane >> 4) & 1) * 8;
#pragma unroll
    for (int r = 0; r < 8; ++r) dst[(mb + r) * ld + n] = c[r] * scale;
}

// one 16x16 output tile over K=128 (4 x wmma_f32_16x16x32_f16)
__device__ __forceinline__ v8f gemm_tile_128(const _Float16* A, int lda,
                                             const _Float16* Bt, int ldb,
                                             int m0, int n0, float bias, int lane) {
    v8f c;
#pragma unroll
    for (int r = 0; r < 8; ++r) c[r] = bias;
#pragma unroll
    for (int kk = 0; kk < 128; kk += 32) {
        v16h a = load_a_frag(A, lda, m0, kk, lane);
        v16h b = load_b_frag_nk(Bt, ldb, kk, n0, lane);
        c = __builtin_amdgcn_wmma_f32_16x16x32_f16(false, a, false, b,
                                                   (short)0, c, false, false);
    }
    return c;
}

// ------------- setup: f32 weights -> transposed f16 in workspace -----------
// dst holds 6 matrices [dout=128][din=128]: Wt[g][j][d] = W[g][d][j]
__global__ void convert_weights_kernel(const float* __restrict__ Wq,
                                       const float* __restrict__ Wk,
                                       const float* __restrict__ Wv,
                                       const float* __restrict__ Wqp,
                                       const float* __restrict__ Wkp,
                                       const float* __restrict__ Wvp,
                                       _Float16* __restrict__ dst) {
    int idx = blockIdx.x * blockDim.x + threadIdx.x;
    if (idx >= 6 * 16384) return;
    int g = idx >> 14, r = idx & 16383, j = r >> 7, d = r & 127;
    const float* W = (g == 0) ? Wq : (g == 1) ? Wk : (g == 2) ? Wv
                    : (g == 3) ? Wqp : (g == 4) ? Wkp : Wvp;
    dst[idx] = (_Float16)W[d * 128 + j];
}

// ----------------- stage A: per-walk attention + Wl linear -----------------
__global__ __launch_bounds__(128) void walk_attn_kernel(
    const float* __restrict__ timef, const float* __restrict__ ppos,
    const float* __restrict__ pneg,  const float* __restrict__ wgt,
    const int* __restrict__ sgn,     const _Float16* __restrict__ wt16,
    const float* __restrict__ bq, const float* __restrict__ bk,
    const float* __restrict__ bv,
    const float* __restrict__ Wl, const float* __restrict__ bl,
    float* __restrict__ emb /* [2][B][NW][M] */) {
    __shared__ _Float16 xh[LW_SZ * F_DIM];
    __shared__ _Float16 qh[LW_SZ * F_DIM];
    __shared__ _Float16 kh[LW_SZ * F_DIM];
    __shared__ _Float16 vh[LW_SZ * F_DIM];
    __shared__ float    sc[LW_SZ * LW_SZ];
    __shared__ float    cw[LW_SZ];
    __shared__ float    ov[F_DIM];

    const int nw = blockIdx.x, b = blockIdx.y, s = blockIdx.z;
    const int tid = threadIdx.x, lane = tid & 31, wave = tid >> 5;
    const long tok0 = ((long)b * NW_SZ + nw) * LW_SZ;

    // build x = [posfeat(19) | time(108) | weight(1)] with sign selection
    for (int idx = tid; idx < LW_SZ * F_DIM; idx += 128) {
        int t = idx >> 7, f = idx & 127;
        long tok = tok0 + t;
        float val;
        if (f < PD_SZ) {
            bool up = (sgn[tok] > 0) != (s == 1);
            val = (up ? ppos : pneg)[tok * PD_SZ + f];
        } else if (f < F_DIM - 1) {
            val = timef[tok * TD_SZ + (f - PD_SZ)];
        } else {
            val = wgt[tok];
        }
        xh[idx] = (_Float16)val;
    }
    __syncthreads();

    // q/k/v : 3 GEMMs of 32x128x128  (48 tiles, 12 per wave)
    for (int tt = wave; tt < 48; tt += 4) {
        int g = tt >> 4, r = tt & 15;
        int m0 = (r >> 3) << 4, n0 = (r & 7) << 4;
        const _Float16* Wt = wt16 + g * 16384;
        const float* bvec = (g == 0) ? bq : (g == 1) ? bk : bv;
        float bias = bvec[n0 + (lane & 15)];
        v8f c = gemm_tile_128(xh, F_DIM, Wt, F_DIM, m0, n0, bias, lane);
        _Float16* dstp = (g == 0) ? qh : (g == 1) ? kh : vh;  // runtime select,
        store_c_f16(dstp, F_DIM, m0, n0, lane, c);            // no static LDS-ptr array
    }
    __syncthreads();

    // scores = q k^T / sqrt(128)   (32x32 = 4 tiles, 1 per wave)
    {
        int m0 = (wave >> 1) << 4, n0 = (wave & 1) << 4;
        v8f c = gemm_tile_128(qh, F_DIM, kh, F_DIM, m0, n0, 0.0f, lane);
        store_c_f32(sc, LW_SZ, m0, n0, lane, c, RSQRT128);
    }
    __syncthreads();

    // softmax rows
    if (tid < LW_SZ) {
        float* row = sc + tid * LW_SZ;
        float mx = row[0];
        for (int j = 1; j < LW_SZ; ++j) mx = fmaxf(mx, row[j]);
        float sum = 0.f;
        for (int j = 0; j < LW_SZ; ++j) { float e = __expf(row[j] - mx); row[j] = e; sum += e; }
        float inv = 1.f / sum;
        for (int j = 0; j < LW_SZ; ++j) row[j] *= inv;
    }
    __syncthreads();
    // column sums:  sum_l w[l][m]   (av.sum over queries == cw @ v)
    if (tid < LW_SZ) {
        float acc = 0.f;
        for (int l = 0; l < LW_SZ; ++l) acc += sc[l * LW_SZ + tid];
        cw[tid] = acc;
    }
    __syncthreads();
    // out[d] = sum_m cw[m]*v[m][d]
    {
        float acc = 0.f;
        for (int m = 0; m < LW_SZ; ++m) acc += cw[m] * (float)vh[m * F_DIM + tid];
        ov[tid] = acc;
    }
    __syncthreads();
    // y = out @ Wl + bl  (coalesced column reads of Wl)
    {
        float acc = bl[tid];
        for (int d = 0; d < F_DIM; ++d) acc += ov[d] * Wl[d * M_DIM + tid];
        emb[(((long)s * B_SZ + b) * NW_SZ + nw) * M_DIM + tid] = acc;
    }
}

// --------------------- stage B: path attention over 64 walks ---------------
__global__ __launch_bounds__(128) void path_attn_kernel(
    const float* __restrict__ emb, const _Float16* __restrict__ wt16,
    const float* __restrict__ bqp, const float* __restrict__ bkp,
    const float* __restrict__ bvp, float* __restrict__ out /* [2][B][M] */) {
    // 64 KB, aliased: xh(16K) <-> sc(64*64 f32); qh(16K) <-> cw(64 f32)
    __shared__ __align__(16) unsigned char smem[4 * NW_SZ * M_DIM * 2];
    _Float16* xh = (_Float16*)smem;
    _Float16* qh = (_Float16*)(smem + 16384);
    _Float16* kh = (_Float16*)(smem + 32768);
    _Float16* vh = (_Float16*)(smem + 49152);
    float* sc = (float*)smem;           // valid after q/k/v done (x dead)
    float* cw = (float*)(smem + 16384); // valid after scores done (q dead)

    const int b = blockIdx.x, s = blockIdx.y;
    const int tid = threadIdx.x, lane = tid & 31, wave = tid >> 5;

    const float* x = emb + ((long)s * B_SZ + b) * NW_SZ * M_DIM;
    for (int idx = tid; idx < NW_SZ * M_DIM; idx += 128)
        xh[idx] = (_Float16)x[idx];
    __syncthreads();

    // q/k/v : 3 GEMMs of 64x128x128  (96 tiles, 24 per wave)
    for (int tt = wave; tt < 96; tt += 4) {
        int g = tt >> 5, r = tt & 31;
        int m0 = (r >> 3) << 4, n0 = (r & 7) << 4;
        const _Float16* Wt = wt16 + (3 + g) * 16384;
        const float* bvec = (g == 0) ? bqp : (g == 1) ? bkp : bvp;
        float bias = bvec[n0 + (lane & 15)];
        v8f c = gemm_tile_128(xh, M_DIM, Wt, M_DIM, m0, n0, bias, lane);
        _Float16* dstp = (g == 0) ? qh : (g == 1) ? kh : vh;
        store_c_f16(dstp, M_DIM, m0, n0, lane, c);
    }
    __syncthreads();

    // scores 64x64 = 16 tiles (4 per wave); sc aliases xh (now dead)
    for (int tt = wave; tt < 16; tt += 4) {
        int m0 = (tt >> 2) << 4, n0 = (tt & 3) << 4;
        v8f c = gemm_tile_128(qh, M_DIM, kh, M_DIM, m0, n0, 0.0f, lane);
        store_c_f32(sc, NW_SZ, m0, n0, lane, c, RSQRT128);
    }
    __syncthreads();

    if (tid < NW_SZ) {
        float* row = sc + tid * NW_SZ;
        float mx = row[0];
        for (int j = 1; j < NW_SZ; ++j) mx = fmaxf(mx, row[j]);
        float sum = 0.f;
        for (int j = 0; j < NW_SZ; ++j) { float e = __expf(row[j] - mx); row[j] = e; sum += e; }
        float inv = 1.f / sum;
        for (int j = 0; j < NW_SZ; ++j) row[j] *= inv;
    }
    __syncthreads();
    if (tid < NW_SZ) {
        float acc = 0.f;
        for (int l = 0; l < NW_SZ; ++l) acc += sc[l * NW_SZ + tid];
        cw[tid] = acc;
    }
    __syncthreads();
    {
        float acc = 0.f;
        for (int m = 0; m < NW_SZ; ++m) acc += cw[m] * (float)vh[m * M_DIM + tid];
        out[((long)s * B_SZ + b) * M_DIM + tid] = acc;
    }
}

// ---------------------------------------------------------------------------
extern "C" void kernel_launch(void* const* d_in, const int* in_sizes, int n_in,
                              void* d_out, int out_size, void* d_ws, size_t ws_size,
                              hipStream_t stream) {
    const float* timef = (const float*)d_in[0];
    const float* ppos  = (const float*)d_in[1];
    const float* pneg  = (const float*)d_in[2];
    const float* wgt   = (const float*)d_in[3];
    const int*   sgn   = (const int*)d_in[4];
    const float* Wq  = (const float*)d_in[5];  const float* bq  = (const float*)d_in[6];
    const float* Wk  = (const float*)d_in[7];  const float* bk  = (const float*)d_in[8];
    const float* Wv  = (const float*)d_in[9];  const float* bv  = (const float*)d_in[10];
    const float* Wl  = (const float*)d_in[11]; const float* bl  = (const float*)d_in[12];
    const float* Wqp = (const float*)d_in[13]; const float* bqp = (const float*)d_in[14];
    const float* Wkp = (const float*)d_in[15]; const float* bkp = (const float*)d_in[16];
    const float* Wvp = (const float*)d_in[17]; const float* bvp = (const float*)d_in[18];

    _Float16* wt16 = (_Float16*)d_ws;                                  // 192 KB
    float* emb = (float*)((char*)d_ws + 6 * 16384 * sizeof(_Float16)); // 8 MB

    convert_weights_kernel<<<(6 * 16384 + 255) / 256, 256, 0, stream>>>(
        Wq, Wk, Wv, Wqp, Wkp, Wvp, wt16);

    dim3 gA(NW_SZ, B_SZ, 2);
    walk_attn_kernel<<<gA, 128, 0, stream>>>(timef, ppos, pneg, wgt, sgn, wt16,
                                             bq, bk, bv, Wl, bl, emb);

    dim3 gB(B_SZ, 2);
    path_attn_kernel<<<gB, 128, 0, stream>>>(emb, wt16, bqp, bkp, bvp,
                                             (float*)d_out);
}
